// BiLSTM_27736898798238
// MI455X (gfx1250) — matrix operand order repaired and checked by reading, not compiled, not run on hardware
//
#include <hip/hip_runtime.h>
#include <hip/hip_bf16.h>
#include <cstdint>
#include <cstddef>

// ---------------------------------------------------------------------------
// BiLSTM (2-layer, bidirectional, H=512, B=32, T=512) for gfx1250 (CDNA5).
// bf16 WMMA (v_wmma_f32_16x16x32_bf16), f32 accumulation.
// Recurrent scan distributed over 8 WGPs/direction with a device-scope
// generation barrier and a ping-pong h buffer in L2.
// ---------------------------------------------------------------------------

typedef __attribute__((ext_vector_type(16))) __bf16 v16bf;
typedef __attribute__((ext_vector_type(8)))  __bf16 v8bf;
typedef __attribute__((ext_vector_type(8)))  float  v8f;

#define T_STEPS 512
#define BATCH   32
#define HID     512
#define G4      2048   // 4*H
#define IN0     128
#define NT_TILES (G4 / 16)   // 128 N-tiles
#define NWG_SCAN 8           // workgroups per direction in the scan

__device__ __forceinline__ float sigm(float x) {
    return 1.0f / (1.0f + __expf(-x));
}

// A-fragment (16x32 bf16) per ISA layout:
//   lanes 0-15 : M=lane,    K = [kb..kb+7] and [kb+16..kb+23]
//   lanes16-31 : M=lane-16, same but kb shifted by +8
// -> two contiguous 16B loads per lane from row-major A.
__device__ __forceinline__ v16bf load_a_frag(const __bf16* row, int kb) {
    v8bf lo = *(const v8bf*)(row + kb);
    v8bf hi = *(const v8bf*)(row + kb + 16);
    return __builtin_shufflevector(lo, hi, 0,1,2,3,4,5,6,7,8,9,10,11,12,13,14,15);
}

// ---------------------------------------------------------------------------
// Pack x [B, I, T] f32 -> xbf [T, B, I] bf16 (row-major A for the layer-0 GEMM)
// ---------------------------------------------------------------------------
__global__ void k_pack_x(const float* __restrict__ x, __bf16* __restrict__ xbf) {
    int idx = blockIdx.x * 256 + threadIdx.x;
    if (idx >= T_STEPS * BATCH * IN0) return;
    int i = idx % IN0;
    int r = idx / IN0;
    int b = r % BATCH;
    int t = r / BATCH;
    xbf[idx] = (__bf16)x[(size_t)b * IN0 * T_STEPS + (size_t)i * T_STEPS + t];
}

// ---------------------------------------------------------------------------
// Pack weight W [N rows, K cols] f32 (logical B[k][n] = W[n*K+k]) into bf16
// WMMA B-fragment tiles. Tile (nt,kt) at (nt*KT + kt)*512 elems; element
// e = lane*16 + v*2 + p  ->  n = nt*16 + (lane&15),
//                            k = kt*32 + (v>=4 ? 16 : 0) + (lane>=16 ? 8 : 0) + (v&3)*2 + p.
// At run time each lane reads its fragment as one contiguous 32B chunk.
// ---------------------------------------------------------------------------
__global__ void k_pack_b(const float* __restrict__ W, __bf16* __restrict__ P,
                         int K, int N) {
    int idx = blockIdx.x * 256 + threadIdx.x;
    if (idx >= K * N) return;
    int tile = idx >> 9;
    int e    = idx & 511;
    int lane = e >> 4;
    int v    = (e >> 1) & 7;
    int pp   = e & 1;
    int KT   = K >> 5;
    int kt   = tile % KT;
    int nt   = tile / KT;
    int n = nt * 16 + (lane & 15);
    int k = kt * 32 + ((v >> 2) << 4) + ((lane >> 4) << 3) + (v & 3) * 2 + pp;
    P[idx] = (__bf16)W[(size_t)n * K + k];
}

// ---------------------------------------------------------------------------
// xp GEMM: Cpacked = A[M,K](bf16) x Bpacked[K,N](bf16) + bias[N].
// Output in C-fragment tile order: [(mtile*NT + ntile)*32 + lane][8 f32]
// so each lane stores (and the scan later loads) one contiguous 32B chunk.
// Block = 256 threads (8 waves); block tile 128x128; wave w owns M-tile w.
// ---------------------------------------------------------------------------
__global__ void __launch_bounds__(256)
k_gemm_xp(const __bf16* __restrict__ A, const __bf16* __restrict__ Bp,
          const float* __restrict__ bias, float* __restrict__ Cp,
          int N, int K) {
    const int lane = threadIdx.x & 31;
    const int wave = threadIdx.x >> 5;
    const int KT = K >> 5;
    const int NT = N >> 4;
    const int n0 = blockIdx.x * 128;
    const int m0 = blockIdx.y * 128 + wave * 16;

    v8f acc[8];
#pragma unroll
    for (int i = 0; i < 8; ++i) acc[i] = v8f{};

    const __bf16* arow = A + (size_t)(m0 + (lane & 15)) * K;
    const int kofs = (lane & 16) ? 8 : 0;
    const int ntbase = n0 >> 4;

    for (int kt = 0; kt < KT; ++kt) {
        v16bf afrag = load_a_frag(arow, kt * 32 + kofs);
#pragma unroll
        for (int nt = 0; nt < 8; ++nt) {
            const __bf16* tb = Bp + ((size_t)((ntbase + nt) * KT + kt) << 9);
            __builtin_prefetch(tb + 512, 0, 0);       // global_prefetch next kt tile
            v16bf bfrag = *((const v16bf*)tb + lane);
            acc[nt] = __builtin_amdgcn_wmma_f32_16x16x32_bf16(
                false, afrag, false, bfrag, (short)0, acc[nt], false, false);
        }
    }

    const int mtile = m0 >> 4;
    const int c15 = lane & 15;
#pragma unroll
    for (int nt = 0; nt < 8; ++nt) {
        const int ntile = ntbase + nt;
        float bv = bias[ntile * 16 + c15];
        v8f r;
#pragma unroll
        for (int i = 0; i < 8; ++i) r[i] = acc[nt][i] + bv;
        *(v8f*)(Cp + (((size_t)mtile * NT + ntile) * 32 + lane) * 8) = r;
    }
}

// ---------------------------------------------------------------------------
// Init for the scan: zero the ping-pong h buffers and the sync counters.
// Must run before every scan launch (graph replays stay deterministic).
// ---------------------------------------------------------------------------
__global__ void k_scan_init(__bf16* __restrict__ hbuf, int* __restrict__ syncc) {
    int idx = blockIdx.x * 256 + threadIdx.x;
    if (idx < 2 * 2 * BATCH * HID) hbuf[idx] = (__bf16)0.0f;
    if (idx < 8) syncc[idx] = 0;
}

// ---------------------------------------------------------------------------
// Recurrent scan, 8 workgroups x 128 threads per direction (32 waves/dir,
// ~1 wave per SIMD across 8 WGPs -> 8x matrix-pipe throughput per step).
// Wave w (0..31) owns hidden units [16w,16w+16): gate N-tiles {g*32+w}.
// h ping-pongs between two global [B,H] bf16 buffers; one device-scope
// generation barrier per step (release fence / atomic arrive / spin /
// acquire fence -> global_inv drops stale L0 lines).
// c and the classifier time-pool live in VGPRs for the whole scan.
// ---------------------------------------------------------------------------
__global__ void __launch_bounds__(128)
k_lstm_scan(const float* __restrict__ xp_f, const float* __restrict__ xp_r,
            const __bf16* __restrict__ WhhP_f, const __bf16* __restrict__ WhhP_r,
            __bf16* __restrict__ hbuf,     // [2 dir][2 phase][B][H] bf16
            int* __restrict__ syncc,       // per-direction counters
            __bf16* __restrict__ hs_out,   // layer 0: [T, B, 2H] bf16
            float* __restrict__ pooled,    // layer 1: [B, 2H] f32 (sum over t)
            int layer) {
    const int lane = threadIdx.x & 31;
    const int wave = threadIdx.x >> 5;
    const int dir  = blockIdx.y;
    const int w    = blockIdx.x * 4 + wave;      // 0..31

    const float*  xp = dir ? xp_r : xp_f;
    const __bf16* Wp = dir ? WhhP_r : WhhP_f;
    int* cnt = syncc + dir;
    __bf16* hb = hbuf + (size_t)dir * 2 * BATCH * HID;

    const int ucol  = w * 16 + (lane & 15);
    const int rbase = (lane & 16) ? 8 : 0;
    const int kofs  = (lane & 16) ? 8 : 0;
    const int arow  = lane & 15;

    float c[2][8], pool[2][8];
#pragma unroll
    for (int mt = 0; mt < 2; ++mt)
#pragma unroll
        for (int r = 0; r < 8; ++r) { c[mt][r] = 0.0f; pool[mt][r] = 0.0f; }

    for (int t = 0; t < T_STEPS; ++t) {
        const int tt = dir ? (T_STEPS - 1 - t) : t;

        // gates init from packed xp (one 32B chunk per tile per lane)
        v8f acc[2][4];
#pragma unroll
        for (int mt = 0; mt < 2; ++mt)
#pragma unroll
            for (int g = 0; g < 4; ++g) {
                size_t mtile = (size_t)tt * 2 + mt;
                size_t ntile = g * 32 + w;
                acc[mt][g] = *(const v8f*)(xp + ((mtile * NT_TILES + ntile) * 32 + lane) * 8);
            }

        // gates += h x WhhPacked  (K = 512, 16 K-steps of 32)
        const __bf16* hread = hb + (size_t)(t & 1) * BATCH * HID;
        for (int kt = 0; kt < 16; ++kt) {
            const int kb = kt * 32 + kofs;
            v16bf a0 = load_a_frag(hread + (size_t)arow * HID, kb);
            v16bf a1 = load_a_frag(hread + (size_t)(16 + arow) * HID, kb);
#pragma unroll
            for (int g = 0; g < 4; ++g) {
                const int ntile = g * 32 + w;
                v16bf bfrag = *((const v16bf*)(Wp + ((size_t)(ntile * 16 + kt) << 9)) + lane);
                acc[0][g] = __builtin_amdgcn_wmma_f32_16x16x32_bf16(
                    false, a0, false, bfrag, (short)0, acc[0][g], false, false);
                acc[1][g] = __builtin_amdgcn_wmma_f32_16x16x32_bf16(
                    false, a1, false, bfrag, (short)0, acc[1][g], false, false);
            }
        }

        // elementwise LSTM cell; write new h slice to the other phase
        __bf16* hwrite = hb + (size_t)((t + 1) & 1) * BATCH * HID;
#pragma unroll
        for (int mt = 0; mt < 2; ++mt)
#pragma unroll
            for (int r = 0; r < 8; ++r) {
                float iv = sigm(acc[mt][0][r]);
                float fv = sigm(acc[mt][1][r]);
                float gv = tanhf(acc[mt][2][r]);
                float ov = sigm(acc[mt][3][r]);
                float cn = fv * c[mt][r] + iv * gv;
                c[mt][r] = cn;
                float hv = ov * tanhf(cn);
                int b = mt * 16 + rbase + r;
                hwrite[(size_t)b * HID + ucol] = (__bf16)hv;
                if (layer == 0)
                    hs_out[((size_t)tt * BATCH + b) * (2 * HID) + dir * HID + ucol] = (__bf16)hv;
                else
                    pool[mt][r] += hv;
            }

        // ---- device-scope generation barrier across this direction's WGs ----
        __builtin_amdgcn_fence(__ATOMIC_RELEASE, "agent");   // flush my h stores
        __syncthreads();
        if (threadIdx.x == 0) {
            __hip_atomic_fetch_add(cnt, 1, __ATOMIC_RELAXED, __HIP_MEMORY_SCOPE_AGENT);
            const int target = (t + 1) * NWG_SCAN;
            while (__hip_atomic_load(cnt, __ATOMIC_RELAXED, __HIP_MEMORY_SCOPE_AGENT) < target)
                __builtin_amdgcn_s_sleep(1);
        }
        __syncthreads();
        __builtin_amdgcn_fence(__ATOMIC_ACQUIRE, "agent");   // invalidate stale L0
    }

    if (layer == 1) {
#pragma unroll
        for (int mt = 0; mt < 2; ++mt)
#pragma unroll
            for (int r = 0; r < 8; ++r) {
                int b = mt * 16 + rbase + r;
                pooled[(size_t)b * (2 * HID) + dir * HID + ucol] = pool[mt][r];
            }
    }
}

// ---------------------------------------------------------------------------
// out[b,o] = mean_t(h2) . Wlin[o,:] + blin[o]
// ---------------------------------------------------------------------------
__global__ void k_linear(const float* __restrict__ pooled,
                         const float* __restrict__ Wlin,
                         const float* __restrict__ blin,
                         float* __restrict__ out) {
    int idx = threadIdx.x;
    if (idx >= BATCH * 10) return;
    int b = idx / 10, o = idx % 10;
    float s = 0.0f;
    for (int j = 0; j < 2 * HID; ++j)
        s += pooled[(size_t)b * (2 * HID) + j] * Wlin[(size_t)o * (2 * HID) + j];
    out[idx] = s * (1.0f / (float)T_STEPS) + blin[o];
}

// ---------------------------------------------------------------------------
extern "C" void kernel_launch(void* const* d_in, const int* in_sizes, int n_in,
                              void* d_out, int out_size, void* d_ws, size_t ws_size,
                              hipStream_t stream) {
    (void)in_sizes; (void)n_in; (void)out_size; (void)ws_size;

    // setup_inputs order: x, then per (layer, dir): Wih, Whh, b; then Wlin, blin
    const float* x     = (const float*)d_in[0];
    const float* Wih0f = (const float*)d_in[1];
    const float* Whh0f = (const float*)d_in[2];
    const float* b0f   = (const float*)d_in[3];
    const float* Wih0r = (const float*)d_in[4];
    const float* Whh0r = (const float*)d_in[5];
    const float* b0r   = (const float*)d_in[6];
    const float* Wih1f = (const float*)d_in[7];
    const float* Whh1f = (const float*)d_in[8];
    const float* b1f   = (const float*)d_in[9];
    const float* Wih1r = (const float*)d_in[10];
    const float* Whh1r = (const float*)d_in[11];
    const float* b1r   = (const float*)d_in[12];
    const float* Wlin  = (const float*)d_in[13];
    const float* blin  = (const float*)d_in[14];
    float* out = (float*)d_out;

    char* ws = (char*)d_ws;
    size_t off = 0;
    auto alloc = [&](size_t bytes) -> void* {
        void* p = ws + off;
        off = (off + bytes + 255) & ~(size_t)255;
        return p;
    };

    __bf16* xbf    = (__bf16*)alloc((size_t)T_STEPS * BATCH * IN0 * 2);
    __bf16* pWih0f = (__bf16*)alloc((size_t)IN0 * G4 * 2);
    __bf16* pWih0r = (__bf16*)alloc((size_t)IN0 * G4 * 2);
    __bf16* pWhh0f = (__bf16*)alloc((size_t)HID * G4 * 2);
    __bf16* pWhh0r = (__bf16*)alloc((size_t)HID * G4 * 2);
    __bf16* pWih1f = (__bf16*)alloc((size_t)(2 * HID) * G4 * 2);
    __bf16* pWih1r = (__bf16*)alloc((size_t)(2 * HID) * G4 * 2);
    __bf16* pWhh1f = (__bf16*)alloc((size_t)HID * G4 * 2);
    __bf16* pWhh1r = (__bf16*)alloc((size_t)HID * G4 * 2);
    float*  xp_f   = (float*)alloc((size_t)T_STEPS * BATCH * G4 * 4);
    float*  xp_r   = (float*)alloc((size_t)T_STEPS * BATCH * G4 * 4);
    __bf16* hs0    = (__bf16*)alloc((size_t)T_STEPS * BATCH * 2 * HID * 2);
    float*  pooled = (float*)alloc((size_t)BATCH * 2 * HID * 4);
    __bf16* hbuf   = (__bf16*)alloc((size_t)2 * 2 * BATCH * HID * 2);
    int*    syncc  = (int*)alloc(8 * sizeof(int));

    // ---- stage 0: convert / pack ----
    {
        int total = T_STEPS * BATCH * IN0;
        k_pack_x<<<(total + 255) / 256, 256, 0, stream>>>(x, xbf);
    }
    auto packB = [&](const float* W, __bf16* P, int K, int N) {
        int total = K * N;
        k_pack_b<<<(total + 255) / 256, 256, 0, stream>>>(W, P, K, N);
    };
    packB(Wih0f, pWih0f, IN0, G4);
    packB(Wih0r, pWih0r, IN0, G4);
    packB(Whh0f, pWhh0f, HID, G4);
    packB(Whh0r, pWhh0r, HID, G4);
    packB(Wih1f, pWih1f, 2 * HID, G4);
    packB(Wih1r, pWih1r, 2 * HID, G4);
    packB(Whh1f, pWhh1f, HID, G4);
    packB(Whh1r, pWhh1r, HID, G4);

    const int M = T_STEPS * BATCH;             // 16384
    dim3 ggrid(G4 / 128, M / 128);             // (16, 128)
    dim3 sgrid(NWG_SCAN, 2);                   // 8 WGs x 2 directions

    // ---- layer 0: input projections + scan ----
    k_gemm_xp<<<ggrid, 256, 0, stream>>>(xbf, pWih0f, b0f, xp_f, G4, IN0);
    k_gemm_xp<<<ggrid, 256, 0, stream>>>(xbf, pWih0r, b0r, xp_r, G4, IN0);
    k_scan_init<<<256, 256, 0, stream>>>(hbuf, syncc);
    k_lstm_scan<<<sgrid, 128, 0, stream>>>(xp_f, xp_r, pWhh0f, pWhh0r,
                                           hbuf, syncc, hs0, pooled, /*layer=*/0);

    // ---- layer 1: input projections (A = hs0, K = 1024) + scan w/ pooling ----
    k_gemm_xp<<<ggrid, 256, 0, stream>>>(hs0, pWih1f, b1f, xp_f, G4, 2 * HID);
    k_gemm_xp<<<ggrid, 256, 0, stream>>>(hs0, pWih1r, b1r, xp_r, G4, 2 * HID);
    k_scan_init<<<256, 256, 0, stream>>>(hbuf, syncc);
    k_lstm_scan<<<sgrid, 128, 0, stream>>>(xp_f, xp_r, pWhh1f, pWhh1r,
                                           hbuf, syncc, hs0 /*unused*/, pooled, /*layer=*/1);

    // ---- classifier ----
    k_linear<<<1, 320, 0, stream>>>(pooled, Wlin, blin, out);
}
